// EquivariantGraphConv_73950746902796
// MI455X (gfx1250) — compile-verified, matrix operand order checked by reading, not run on hardware
//
#include <hip/hip_runtime.h>
#include <hip/hip_bf16.h>
#include <cstdint>

// ---------------------------------------------------------------------------
// EGNN layer, fused for MI455X (gfx1250, wave32, WMMA bf16 16x16x32).
// B=2, N=512, D=HID=OUT=128.
// ---------------------------------------------------------------------------

typedef __attribute__((ext_vector_type(16))) __bf16 v16bf;
typedef __attribute__((ext_vector_type(8)))  float  v8f;

#define BATCH   2
#define NPTS    512
#define DIM     128
#define NCHUNK  16          // j split into 16 chunks of 32
#define JT      32

// workspace byte offsets (total ~9.3 MB)
#define OFF_PI   (0)                         // B*N*128 f32   = 524288
#define OFF_PJ   (524288)                    // B*N*128 f32   = 524288
#define OFF_WD   (1048576)                   // 128 f32 (pad 1024)
#define OFF_FRAG (1049600)                   // 2*8*4*32*16 bf16 = 65536
#define OFF_AGG  (1115136)                   // NCHUNK*B*N*128 f32 = 8388608
#define OFF_CP   (9503744)                   // NCHUNK*B*N*3 f32   = 196608

// dynamic-LDS partition (bytes) for the edge kernel, 4 waves / 128 threads
#define SM_W     (0)            // weight frags bf16: 2*8*4*32*16*2 = 65536
#define SM_PJ    (65536)        // Pj tile f32: 32*128*4            = 16384
#define SM_PI    (81920)        // Pi tiles f32 padded: 4*16*132*4  = 33792
#define SM_SCR   (115712)       // edge_hidden scratch bf16: 4*16*136*2 = 17408
#define SM_WD    (133120)       // 128 f32 = 512
#define SM_CJ    (133632)       // 32*4 f32 = 512
#define SM_CI    (134144)       // 64*4 f32 = 1024
#define SM_TOTAL (135168)       // ~132 KB (<160 KB/CU, ISA allows 320 KB/WG)

// silu via v_exp_f32 + v_rcp_f32 (no IEEE div sequence)
__device__ __forceinline__ float siluf(float x) {
    return x * __builtin_amdgcn_rcpf(1.0f + __expf(-x));
}

// K index within a 16-bit A/B fragment half-pair (ISA 7.12.2, 16-bit A 16x32):
// element e (pair p=e/2): e<8 -> K = e + (hi?8:0); e>=8 -> K = e + (hi?16:8)
__device__ __forceinline__ int kbase(int p, int hi) {
    return (p < 4) ? (2 * p + (hi ? 8 : 0)) : (2 * p + (hi ? 16 : 8));
}

// ---------------------------------------------------------------------------
// prep 1: Pi = F @ eW1[0:128] + eb1 ; Pj = F @ eW1[128:256]; copy w_d row.
// grid = B*N blocks, 128 threads
// ---------------------------------------------------------------------------
__global__ void egnn_prep_pipj(const float* __restrict__ features,
                               const float* __restrict__ eW1,
                               const float* __restrict__ eb1,
                               float* __restrict__ Pi, float* __restrict__ Pj,
                               float* __restrict__ wd) {
    const int row = blockIdx.x;          // 0 .. B*N-1
    const int c   = threadIdx.x;         // 0 .. 127
    __shared__ float sF[DIM];
    sF[c] = features[(size_t)row * DIM + c];
    __syncthreads();
    float pi = eb1[c];
    float pj = 0.0f;
    #pragma unroll 4
    for (int k = 0; k < DIM; ++k) {
        const float f = sF[k];
        pi += f * eW1[k * DIM + c];
        pj += f * eW1[(DIM + k) * DIM + c];
    }
    Pi[(size_t)row * DIM + c] = pi;
    Pj[(size_t)row * DIM + c] = pj;
    if (row == 0) wd[c] = eW1[2 * DIM * DIM + c];   // distance weight row
}

// ---------------------------------------------------------------------------
// prep 2: pack eW2 / cW1 into bf16 WMMA B-fragment order:
// frag[mat][n][kk][lane][e], value = W[K(e,hi)+32*kk][16*n + lane%16]
// grid = 64 blocks (mat*32 + n*4 + kk), 512 threads (lane*16 + e)
// ---------------------------------------------------------------------------
__global__ void egnn_prep_frags(const float* __restrict__ eW2,
                                const float* __restrict__ cW1,
                                __bf16* __restrict__ frag) {
    const int blk  = blockIdx.x;
    const int mat  = blk >> 5;
    const int n    = (blk >> 2) & 7;
    const int kk   = blk & 3;
    const int lane = threadIdx.x >> 4;
    const int e    = threadIdx.x & 15;
    const int hi   = lane >> 4;
    const int K    = kk * 32 + kbase(e >> 1, hi) + (e & 1);
    const int col  = n * 16 + (lane & 15);
    const float* W = mat ? cW1 : eW2;
    const float v  = W[K * DIM + col];
    frag[((((mat * 8 + n) * 4 + kk) * 32 + lane) * 16) + e] = (__bf16)v;
}

// ---------------------------------------------------------------------------
// main fused edge kernel.
// grid = (NCHUNK, N/64, B), block = 128 threads (4 waves, 16 i-rows each).
// Weight fragments live in LDS (staged once per workgroup) so inner-loop
// B-operands are cheap ds_load_b128s -> no global-load hoisting, no spills.
// Per wave, per j:
//   build A = silu(Pi[i]+Pj[j]+d*wd) bf16 frags -> GEMM1 (eW2) -> agg +=,
//   LDS roundtrip -> GEMM2 (cW1) -> silu -> dot cW2 -> shfl-reduce ->
//   coord-weight accumulation (factored: sum cw and sum cw*cj).
// ---------------------------------------------------------------------------
__global__ void __launch_bounds__(128)
egnn_edge_kernel(const float* __restrict__ coords,
                 const float* __restrict__ eb2,
                 const float* __restrict__ cb1,
                 const float* __restrict__ cW2,
                 const float* __restrict__ cb2,
                 const float* __restrict__ PiG,
                 const float* __restrict__ PjG,
                 const float* __restrict__ wdG,
                 const __bf16* __restrict__ fragG,
                 float* __restrict__ aggP,
                 float* __restrict__ cpP) {
    extern __shared__ char smem[];
    __bf16* sW   = (__bf16*)(smem + SM_W);
    float*  sPj  = (float*)(smem + SM_PJ);
    float*  sPi  = (float*)(smem + SM_PI);
    __bf16* sScr = (__bf16*)(smem + SM_SCR);
    float*  sWd  = (float*)(smem + SM_WD);
    float*  sCj  = (float*)(smem + SM_CJ);
    float*  sCi  = (float*)(smem + SM_CI);

    const int t    = threadIdx.x;
    const int wv   = t >> 5;
    const int lane = t & 31;
    const int lm   = lane & 15;
    const int hi   = lane >> 4;
    const int cx   = blockIdx.x;            // j chunk
    const int ig   = blockIdx.y;            // i-group of 64
    const int b    = blockIdx.z;
    const int jbase = cx * JT;
    const int ibase = ig * 64;

    // ---- cooperative staging ----
    // weight fragments: 65536 B = 4096 uint4
    for (int idx = t; idx < 4096; idx += 128)
        ((uint4*)sW)[idx] = ((const uint4*)fragG)[idx];
    // Pj tile: 4096 floats = 1024 float4
    for (int idx = t; idx < JT * DIM / 4; idx += 128)
        ((float4*)sPj)[idx] =
            ((const float4*)(PjG + (size_t)(b * NPTS + jbase) * DIM))[idx];
    // Pi tiles (padded stride 132)
    for (int idx = t; idx < 4 * 16 * DIM; idx += 128) {
        const int w2 = idx >> 11, rm = (idx >> 7) & 15, c = idx & 127;
        sPi[(w2 * 16 + rm) * 132 + c] =
            PiG[(size_t)(b * NPTS + ibase + w2 * 16 + rm) * DIM + c];
    }
    if (t < DIM) sWd[t] = wdG[t];
    if (t < 32) {
        #pragma unroll
        for (int k = 0; k < 3; ++k)
            sCj[t * 4 + k] = coords[(size_t)(b * NPTS + jbase + t) * 3 + k];
    } else if (t < 96) {
        const int tt = t - 32;
        #pragma unroll
        for (int k = 0; k < 3; ++k)
            sCi[tt * 4 + k] = coords[(size_t)(b * NPTS + ibase + tt) * 3 + k];
    }
    __syncthreads();

    // per-lane constants
    float eb2v[8], cb1v[8], cw2v[8];
    #pragma unroll
    for (int n = 0; n < 8; ++n) {
        eb2v[n] = eb2[n * 16 + lm];
        cb1v[n] = cb1[n * 16 + lm];
        cw2v[n] = cW2[n * 16 + lm];
    }
    const float cb2s = cb2[0];
    const float cix = sCi[(wv * 16 + lm) * 4 + 0];
    const float ciy = sCi[(wv * 16 + lm) * 4 + 1];
    const float ciz = sCi[(wv * 16 + lm) * 4 + 2];

    v8f agg[8];
    #pragma unroll
    for (int n = 0; n < 8; ++n)
        #pragma unroll
        for (int r = 0; r < 8; ++r) agg[n][r] = 0.0f;
    // coord-weight partials: sum(cw) and sum(cw * c_j) per output row r
    float s0[8] = {0, 0, 0, 0, 0, 0, 0, 0};
    float sx[8] = {0, 0, 0, 0, 0, 0, 0, 0};
    float sy[8] = {0, 0, 0, 0, 0, 0, 0, 0};
    float sz[8] = {0, 0, 0, 0, 0, 0, 0, 0};

    const float*  sPiW  = sPi  + wv * 16 * 132;
    __bf16*       sScrW = sScr + wv * 16 * 136;

    #pragma unroll 1
    for (int jj = 0; jj < JT; ++jj) {
        const float cjx = sCj[jj * 4 + 0];
        const float cjy = sCj[jj * 4 + 1];
        const float cjz = sCj[jj * 4 + 2];
        const float dx = cix - cjx, dy = ciy - cjy, dz = ciz - cjz;
        const float dist = sqrtf(dx * dx + dy * dy + dz * dz + 1e-8f);
        const float* pjRow = sPj + jj * DIM;

        // ---- build A fragments: silu(h1) in bf16, ISA A-layout ----
        v16bf A[4];
        #pragma unroll
        for (int kk = 0; kk < 4; ++kk) {
            #pragma unroll
            for (int p = 0; p < 8; ++p) {
                const int K  = kk * 32 + kbase(p, hi);
                const float2 pi2 = *(const float2*)&sPiW[lm * 132 + K];
                const float2 pj2 = *(const float2*)&pjRow[K];
                const float2 wd2 = *(const float2*)&sWd[K];
                const float a0 = siluf(pi2.x + pj2.x + dist * wd2.x);
                const float a1 = siluf(pi2.y + pj2.y + dist * wd2.y);
                A[kk][2 * p]     = (__bf16)a0;
                A[kk][2 * p + 1] = (__bf16)a1;
            }
        }

        // ---- GEMM1: edge_hidden = silu(h1) @ eW2 + eb2 ----
        #pragma unroll
        for (int n = 0; n < 8; ++n) {
            v8f c;
            #pragma unroll
            for (int r = 0; r < 8; ++r) c[r] = eb2v[n];
            #pragma unroll
            for (int kk = 0; kk < 4; ++kk) {
                const v16bf Bf =
                    *(const v16bf*)(sW + (((n * 4 + kk) * 32 + lane) * 16));
                c = __builtin_amdgcn_wmma_f32_16x16x32_bf16(
                        false, A[kk], false, Bf, (short)0, c, false, false);
            }
            #pragma unroll
            for (int r = 0; r < 8; ++r) {
                agg[n][r] += c[r];
                sScrW[(r + 8 * hi) * 136 + (n * 16 + lm)] = (__bf16)c[r];
            }
        }

        // wait for LDS scratch writes before the cross-lane re-read
        asm volatile("s_wait_dscnt 0" ::: "memory");

        // ---- reload edge_hidden tile as A fragments (C-layout -> A-layout) --
        v16bf A2[4];
        #pragma unroll
        for (int kk = 0; kk < 4; ++kk) {
            #pragma unroll
            for (int p = 0; p < 8; ++p) {
                const int K = kk * 32 + kbase(p, hi);
                A2[kk][2 * p]     = sScrW[lm * 136 + K];
                A2[kk][2 * p + 1] = sScrW[lm * 136 + K + 1];
            }
        }

        // ---- GEMM2: t = silu(edge_hidden @ cW1 + cb1); cw = t . cW2 + cb2 --
        float cwr[8] = {0, 0, 0, 0, 0, 0, 0, 0};
        #pragma unroll
        for (int n = 0; n < 8; ++n) {
            v8f c;
            #pragma unroll
            for (int r = 0; r < 8; ++r) c[r] = cb1v[n];
            #pragma unroll
            for (int kk = 0; kk < 4; ++kk) {
                const v16bf Bf = *(const v16bf*)(
                    sW + ((((8 + n) * 4 + kk) * 32 + lane) * 16));
                c = __builtin_amdgcn_wmma_f32_16x16x32_bf16(
                        false, A2[kk], false, Bf, (short)0, c, false, false);
            }
            #pragma unroll
            for (int r = 0; r < 8; ++r)
                cwr[r] += siluf(c[r]) * cw2v[n];
        }

        // reduce across the 16-lane N-group; accumulate cw sums with the
        // wave-uniform c_j (coord update factored as ci*sum(cw) - sum(cw*cj))
        #pragma unroll
        for (int r = 0; r < 8; ++r) {
            float v = cwr[r];
            v += __shfl_xor(v, 1);
            v += __shfl_xor(v, 2);
            v += __shfl_xor(v, 4);
            v += __shfl_xor(v, 8);
            v += cb2s;                       // cw for row M = r + 8*hi
            s0[r] += v;
            sx[r] += v * cjx;
            sy[r] += v * cjy;
            sz[r] += v * cjz;
        }
    }

    // ---- write partial sums ----
    const int i0 = ibase + wv * 16;
    float* aggOut = aggP + (size_t)((cx * BATCH + b) * NPTS + i0) * DIM;
    #pragma unroll
    for (int n = 0; n < 8; ++n)
        #pragma unroll
        for (int r = 0; r < 8; ++r)
            aggOut[(r + 8 * hi) * DIM + n * 16 + lm] = agg[n][r];

    if (lm == 0) {
        float* cpOut = cpP + (size_t)((cx * BATCH + b) * NPTS + i0) * 3;
        #pragma unroll
        for (int r = 0; r < 8; ++r) {
            const int M = r + 8 * hi;
            const float mx = sCi[(wv * 16 + M) * 4 + 0];
            const float my = sCi[(wv * 16 + M) * 4 + 1];
            const float mz = sCi[(wv * 16 + M) * 4 + 2];
            cpOut[M * 3 + 0] = mx * s0[r] - sx[r];
            cpOut[M * 3 + 1] = my * s0[r] - sy[r];
            cpOut[M * 3 + 2] = mz * s0[r] - sz[r];
        }
    }
}

// ---------------------------------------------------------------------------
// reduce chunks + node MLP + coord update. grid = B*N blocks, 128 threads.
// d_out: [B*N*128 updated_features][B*N*3 updated_coords]
// ---------------------------------------------------------------------------
__global__ void egnn_reduce_node(const float* __restrict__ features,
                                 const float* __restrict__ coords,
                                 const float* __restrict__ nW1,
                                 const float* __restrict__ nb1,
                                 const float* __restrict__ nW2,
                                 const float* __restrict__ nb2,
                                 const float* __restrict__ aggP,
                                 const float* __restrict__ cpP,
                                 float* __restrict__ out) {
    const int row = blockIdx.x;            // b*N + n
    const int b   = row >> 9;
    const int nn  = row & (NPTS - 1);
    const int c   = threadIdx.x;

    float s = 0.0f;
    #pragma unroll
    for (int ch = 0; ch < NCHUNK; ++ch)
        s += aggP[(size_t)((ch * BATCH + b) * NPTS + nn) * DIM + c];
    const float aggv = s * (1.0f / (float)NPTS);

    __shared__ float sIn[2 * DIM];
    __shared__ float sH[DIM];
    sIn[c]       = features[(size_t)row * DIM + c];
    sIn[DIM + c] = aggv;
    __syncthreads();

    float h = nb1[c];
    #pragma unroll 4
    for (int k = 0; k < 2 * DIM; ++k) h += sIn[k] * nW1[k * DIM + c];
    sH[c] = siluf(h);
    __syncthreads();

    float o = nb2[c];
    #pragma unroll 4
    for (int k = 0; k < DIM; ++k) o += sH[k] * nW2[k * DIM + c];
    out[(size_t)row * DIM + c] = o;

    if (c < 3) {
        float cs = 0.0f;
        #pragma unroll
        for (int ch = 0; ch < NCHUNK; ++ch)
            cs += cpP[(size_t)((ch * BATCH + b) * NPTS + nn) * 3 + c];
        out[(size_t)BATCH * NPTS * DIM + (size_t)row * 3 + c] =
            coords[(size_t)row * 3 + c] + cs * (1.0f / (float)NPTS);
    }
}

// ---------------------------------------------------------------------------
extern "C" void kernel_launch(void* const* d_in, const int* in_sizes, int n_in,
                              void* d_out, int out_size, void* d_ws,
                              size_t ws_size, hipStream_t stream) {
    const float* features = (const float*)d_in[0];
    const float* coords   = (const float*)d_in[1];
    const float* eW1      = (const float*)d_in[2];
    const float* eb1      = (const float*)d_in[3];
    const float* eW2      = (const float*)d_in[4];
    const float* eb2      = (const float*)d_in[5];
    const float* nW1      = (const float*)d_in[6];
    const float* nb1      = (const float*)d_in[7];
    const float* nW2      = (const float*)d_in[8];
    const float* nb2      = (const float*)d_in[9];
    const float* cW1      = (const float*)d_in[10];
    const float* cb1      = (const float*)d_in[11];
    const float* cW2      = (const float*)d_in[12];
    const float* cb2      = (const float*)d_in[13];

    char* ws = (char*)d_ws;
    float*  Pi   = (float*)(ws + OFF_PI);
    float*  Pj   = (float*)(ws + OFF_PJ);
    float*  wd   = (float*)(ws + OFF_WD);
    __bf16* frag = (__bf16*)(ws + OFF_FRAG);
    float*  aggP = (float*)(ws + OFF_AGG);
    float*  cpP  = (float*)(ws + OFF_CP);

    egnn_prep_pipj<<<BATCH * NPTS, DIM, 0, stream>>>(features, eW1, eb1,
                                                     Pi, Pj, wd);
    egnn_prep_frags<<<64, 512, 0, stream>>>(eW2, cW1, frag);
    egnn_edge_kernel<<<dim3(NCHUNK, NPTS / 64, BATCH), 128, SM_TOTAL, stream>>>(
        coords, eb2, cb1, cW2, cb2, Pi, Pj, wd, frag, aggP, cpP);
    egnn_reduce_node<<<BATCH * NPTS, DIM, 0, stream>>>(
        features, coords, nW1, nb1, nW2, nb2, aggP, cpP, (float*)d_out);
}
